// GPS_34394098106860
// MI455X (gfx1250) — matrix operand order repaired and checked by reading, not compiled, not run on hardware
//
#include <hip/hip_runtime.h>
#include <hip/hip_bf16.h>
#include <stdint.h>

// ---------------------------------------------------------------------------
// Problem constants (match reference)
// ---------------------------------------------------------------------------
static const int Nn  = 65536;   // nodes
static const int Dd  = 256;     // hidden dim
static const int Ee  = 524288;  // edges
static const int Gg  = 512;     // graphs
static const int NPG = 128;     // nodes per graph
static const int Ll  = 4;       // layers
static const int FFf = 512;     // MLP hidden

typedef __attribute__((ext_vector_type(16))) __bf16 v16bf;
typedef __attribute__((ext_vector_type(8)))  float  v8f;

// Types for the gfx1250 async global->LDS builtin:
//   arg0: global (addrspace 1) int4*   arg1: LDS (addrspace 3) int4*
typedef int v4i_ __attribute__((__vector_size__(16)));
typedef __attribute__((address_space(1))) v4i_* gv4_t;
typedef __attribute__((address_space(3))) v4i_* lv4_t;

// ---------------------------------------------------------------------------
// f32 -> bf16 via native conversion
// ---------------------------------------------------------------------------
__device__ __forceinline__ __bf16 f2bf(float f) { return (__bf16)f; }

// ---------------------------------------------------------------------------
// Async global->LDS staging (CDNA5 ASYNCcnt path)
// ---------------------------------------------------------------------------
#if __has_builtin(__builtin_amdgcn_global_load_async_to_lds_b128)
#define HAVE_ASYNC_LDS 1
#else
#define HAVE_ASYNC_LDS 0
#endif

__device__ __forceinline__ void wait_async_le(int n) {
#if HAVE_ASYNC_LDS
#if __has_builtin(__builtin_amdgcn_s_wait_asynccnt)
  if (n == 0) __builtin_amdgcn_s_wait_asynccnt(0);   // builtin needs ICE arg
  else        __builtin_amdgcn_s_wait_asynccnt(2);
#else
  if (n == 0) asm volatile("s_wait_asynccnt 0x0" ::: "memory");
  else        asm volatile("s_wait_asynccnt 0x2" ::: "memory");
#endif
#else
  (void)n;
#endif
}

// Stage a 32 x 64 f32 tile of W (row-major, stride M) into LDS (stride 64).
// 8 KB => 512 x b128; 256 threads issue 2 async b128 copies each.
__device__ __forceinline__ void stage_b_tile(const float* W, int M, int k0,
                                             int col0, float* lbuf, int tid) {
#if HAVE_ASYNC_LDS
#pragma unroll
  for (int u = 0; u < 2; ++u) {
    int q   = tid + u * 256;        // float4 index within tile
    int row = q >> 4;
    int c4  = (q & 15) << 2;
    const float* gp = W + (size_t)(k0 + row) * M + col0 + c4;
    __builtin_amdgcn_global_load_async_to_lds_b128(
        (gv4_t)(void*)(float*)gp, (lv4_t)(void*)(lbuf + row * 64 + c4), 0, 0);
  }
#else
#pragma unroll
  for (int u = 0; u < 2; ++u) {
    int q   = tid + u * 256;
    int row = q >> 4;
    int c4  = (q & 15) << 2;
    const float4 v = *(const float4*)(W + (size_t)(k0 + row) * M + col0 + c4);
    *(float4*)(lbuf + row * 64 + c4) = v;
  }
#endif
}

// ---------------------------------------------------------------------------
// WMMA fragment loaders (CDNA5 wave32 layouts, CDNA5 ISA 7.12.2)
// ---------------------------------------------------------------------------
__device__ __forceinline__ v16bf load_a_frag(const float* A, int lda,
                                             int row0, int k0, int lane) {
  const int half = (lane >> 4) & 1, r = lane & 15;
  const float* ap = A + (size_t)(row0 + r) * lda + k0 + half * 16;
  v16bf out;
#pragma unroll
  for (int v = 0; v < 8; ++v) {
    int k = 2 * v;
    out[2 * v]     = f2bf(ap[k]);
    out[2 * v + 1] = f2bf(ap[k + 1]);
  }
  return out;
}

// B 32x16, source row-major [K x M]: lane column n = n0 + (lane&15),
// lanes 0-15 hold K=k0..k0+15 (2 per VGPR), lanes 16-31 hold K=k0+16..k0+31.
__device__ __forceinline__ v16bf load_b_frag(const float* B, int ldb,
                                             int k0, int n0, int lane) {
  const int half = (lane >> 4) & 1, r = lane & 15;
  const float* bp = B + (size_t)(k0 + half * 16) * ldb + n0 + r;
  v16bf out;
#pragma unroll
  for (int v = 0; v < 8; ++v) {
    out[2 * v]     = f2bf(bp[(size_t)(2 * v) * ldb]);
    out[2 * v + 1] = f2bf(bp[(size_t)(2 * v + 1) * ldb]);
  }
  return out;
}

// B fragment where logical B[k][n] lives at P[n*ld + k] (K^T in attention).
__device__ __forceinline__ v16bf load_bt_frag(const float* P, int ld,
                                              int k0, int n0, int lane) {
  const int half = (lane >> 4) & 1, r = lane & 15;
  const float* bp = P + (size_t)(n0 + r) * ld + k0 + half * 16;
  v16bf out;
#pragma unroll
  for (int v = 0; v < 8; ++v) {
    out[2 * v]     = f2bf(bp[2 * v]);
    out[2 * v + 1] = f2bf(bp[2 * v + 1]);
  }
  return out;
}

__device__ __forceinline__ v8f wmma_bf16(v16bf a, v16bf b, v8f c) {
  return __builtin_amdgcn_wmma_f32_16x16x32_bf16(false, a, false, b,
                                                 (short)0, c, false, false);
}

// ---------------------------------------------------------------------------
// Generic GEMM: Y[N x M] = epilogue(A[N x K] @ W[K x M] + bias, R)
// mode 0: +bias  1: relu(+bias)  2: relu(+bias)+R  3: +bias+R
// Block = 256 thr (8 waves); wave tile 32x32; block tile 128x64.
// B tile staged to LDS (double-buffered, async ASYNCcnt DMA) so the 4
// row-wave groups share one copy of the weight tile.
// ---------------------------------------------------------------------------
__global__ __launch_bounds__(256) void wmma_gemm_kernel(
    const float* __restrict__ A, const float* __restrict__ W,
    const float* __restrict__ bias, const float* __restrict__ R,
    float* __restrict__ Y, int K, int M, int mode) {
  __shared__ float lbuf[2][32 * 64];             // 16 KB double buffer
  const int tid  = threadIdx.x;
  const int lane = tid & 31;
  const int wid  = tid >> 5;
  const int rowBase = blockIdx.x * 128 + (wid & 3) * 32;
  const int colBlk  = blockIdx.y * 64;
  const int colLoc  = (wid >> 2) * 32;
  v8f acc[2][2] = {};
  const int nk = K >> 5;

  stage_b_tile(W, M, 0, colBlk, lbuf[0], tid);
  for (int ks = 0; ks < nk; ++ks) {
    const int k0 = ks * 32;
    if (ks + 1 < nk)
      stage_b_tile(W, M, k0 + 32, colBlk, lbuf[(ks + 1) & 1], tid);
    wait_async_le(ks + 1 < nk ? 2 : 0);          // stage ks complete (in-order)
    __syncthreads();                             // visible to all waves
    if (k0 + 32 < K)                             // next A tile prefetch hint
      __builtin_prefetch(A + (size_t)(rowBase + (lane & 15)) * K + k0 + 32, 0, 1);
    const float* bt = lbuf[ks & 1];
    v16bf a0 = load_a_frag(A, K, rowBase,      k0, lane);
    v16bf a1 = load_a_frag(A, K, rowBase + 16, k0, lane);
    v16bf b0 = load_b_frag(bt, 64, 0, colLoc,      lane);
    v16bf b1 = load_b_frag(bt, 64, 0, colLoc + 16, lane);
    acc[0][0] = wmma_bf16(a0, b0, acc[0][0]);
    acc[0][1] = wmma_bf16(a0, b1, acc[0][1]);
    acc[1][0] = wmma_bf16(a1, b0, acc[1][0]);
    acc[1][1] = wmma_bf16(a1, b1, acc[1][1]);
    __syncthreads();                             // done reading lbuf[ks&1]
  }

  const int half = (lane >> 4) & 1, r = lane & 15;
#pragma unroll
  for (int i = 0; i < 2; ++i)
#pragma unroll
    for (int j = 0; j < 2; ++j)
#pragma unroll
      for (int p = 0; p < 8; ++p) {
        int row = rowBase + 16 * i + p + 8 * half;
        int col = colBlk + colLoc + 16 * j + r;
        float y = acc[i][j][p] + bias[col];
        if (mode == 1 || mode == 2) y = fmaxf(y, 0.f);
        if (mode == 2 || mode == 3) y += R[(size_t)row * M + col];
        Y[(size_t)row * M + col] = y;
      }
}

// ---------------------------------------------------------------------------
// Dense per-graph multi-head attention. One block per (graph, head).
// scores = softmax(Q K^T / sqrt(DH)); o-slice = scores @ V. All matmuls WMMA.
// ---------------------------------------------------------------------------
__global__ __launch_bounds__(256) void attn_kernel(const float* __restrict__ qkv,
                                                   float* __restrict__ o) {
  __shared__ float sc[NPG * NPG];                 // 64 KB score matrix
  const int g    = blockIdx.x >> 3;               // / H
  const int hh   = blockIdx.x & 7;
  const int lane = threadIdx.x & 31;
  const int wid  = threadIdx.x >> 5;
  const float* qg = qkv + (size_t)g * NPG * 768;  // qkv row stride = 3*D
  const float* Qb = qg + hh * 32;
  const float* Kb = qg + 256 + hh * 32;
  const float* Vb = qg + 512 + hh * 32;
  const int rows0 = wid * 16;                     // 8 waves x 16 rows = 128
  const int half = (lane >> 4) & 1, r = lane & 15;
  const float scale = 0.17677669529663688f;       // 1/sqrt(32)

  // ---- scores: Q (16x32) @ K^T (32x16 per tile), single K step ----
  v16bf qa = load_a_frag(Qb, 768, rows0, 0, lane);
#pragma unroll
  for (int ct = 0; ct < 8; ++ct) {
    v16bf kb = load_bt_frag(Kb, 768, 0, ct * 16, lane);
    v8f acc = {};
    acc = wmma_bf16(qa, kb, acc);
#pragma unroll
    for (int p = 0; p < 8; ++p) {
      int row = rows0 + p + 8 * half;
      int col = ct * 16 + r;
      sc[row * NPG + col] = acc[p] * scale;
    }
  }
  __syncthreads();

  // ---- row softmax (128 rows, one per thread) ----
  if (threadIdx.x < NPG) {
    float* rp = &sc[threadIdx.x * NPG];
    float m = rp[0];
    for (int i = 1; i < NPG; ++i) m = fmaxf(m, rp[i]);
    float s = 0.f;
    for (int i = 0; i < NPG; ++i) { float e = __expf(rp[i] - m); rp[i] = e; s += e; }
    float inv = 1.f / s;
    for (int i = 0; i < NPG; ++i) rp[i] *= inv;
  }
  __syncthreads();

  // ---- attn (16x128) @ V (128x32): 4 K-steps, 2 col tiles ----
  v8f acc2[2] = {};
#pragma unroll
  for (int ks = 0; ks < 4; ++ks) {
    v16bf aa = load_a_frag(sc, NPG, rows0, ks * 32, lane);
    v16bf v0 = load_b_frag(Vb, 768, ks * 32, 0,  lane);
    v16bf v1 = load_b_frag(Vb, 768, ks * 32, 16, lane);
    acc2[0] = wmma_bf16(aa, v0, acc2[0]);
    acc2[1] = wmma_bf16(aa, v1, acc2[1]);
  }
#pragma unroll
  for (int j = 0; j < 2; ++j)
#pragma unroll
    for (int p = 0; p < 8; ++p) {
      int row = g * NPG + rows0 + p + 8 * half;
      int col = hh * 32 + j * 16 + r;
      o[(size_t)row * Dd + col] = acc2[j][p];
    }
}

// ---------------------------------------------------------------------------
// Node encoder: h = sum_c atom_emb[c, x[n,c]] + pe @ pe_w + pe_b
// ---------------------------------------------------------------------------
__global__ __launch_bounds__(256) void encode_kernel(
    const int* __restrict__ x, const float* __restrict__ pe,
    const float* __restrict__ atom_emb, const float* __restrict__ pe_w,
    const float* __restrict__ pe_b, float* __restrict__ h) {
  const int n = blockIdx.x, d = threadIdx.x;
  float s = pe_b[d];
#pragma unroll
  for (int c = 0; c < 9; ++c) {
    int xi = x[n * 9 + c];                               // uniform per block
    s += atom_emb[((size_t)c * 128 + xi) * 256 + d];
  }
#pragma unroll
  for (int w = 0; w < 20; ++w)
    s += pe[(size_t)n * 20 + w] * pe_w[w * 256 + d];
  h[(size_t)n * 256 + d] = s;
}

// ---------------------------------------------------------------------------
// GINE message: m = relu(h[src] + bond_emb_sum(edge)); scatter-add to agg[dst]
// ---------------------------------------------------------------------------
__global__ __launch_bounds__(256) void message_kernel(
    const float* __restrict__ h, const int* __restrict__ src,
    const int* __restrict__ dst, const int* __restrict__ edge_attr,
    const float* __restrict__ bond_emb, float* __restrict__ agg) {
  const int e = blockIdx.x, d = threadIdx.x;
  const int s = src[e], t = dst[e];                      // uniform per block
  float ea = 0.f;
#pragma unroll
  for (int c = 0; c < 3; ++c) {
    int bi = edge_attr[e * 3 + c];
    ea += bond_emb[((size_t)c * 8 + bi) * 256 + d];
  }
  float m = fmaxf(h[(size_t)s * 256 + d] + ea, 0.f);
  atomicAdd(&agg[(size_t)t * 256 + d], m);
}

__global__ void add_kernel(float* __restrict__ a, const float* __restrict__ b,
                           size_t n) {
  size_t i = (size_t)blockIdx.x * blockDim.x + threadIdx.x;
  if (i < n) a[i] += b[i];
}

// ---------------------------------------------------------------------------
// BatchNorm over node axis: two-pass (column sums/sumsq via atomics, apply)
// ---------------------------------------------------------------------------
__global__ __launch_bounds__(256) void bn_stats_kernel(const float* __restrict__ T,
                                                       float* __restrict__ stats) {
  const int d = threadIdx.x;
  const int rowBase = blockIdx.x * 256;
  float s = 0.f, ss = 0.f;
  for (int rr = 0; rr < 256; ++rr) {
    float v = T[(size_t)(rowBase + rr) * 256 + d];
    s += v; ss += v * v;
  }
  atomicAdd(&stats[d], s);
  atomicAdd(&stats[256 + d], ss);
}

__global__ __launch_bounds__(256) void bn_apply_kernel(
    const float* __restrict__ T, const float* __restrict__ stats,
    const float* __restrict__ g, const float* __restrict__ b,
    float* __restrict__ Y, int relu) {
  const int d = threadIdx.x;
  const size_t n = blockIdx.x;
  const float inv_n = 1.f / 65536.f;
  float mu  = stats[d] * inv_n;
  float var = stats[256 + d] * inv_n - mu * mu;
  float v = (T[n * 256 + d] - mu) * rsqrtf(var + 1e-5f) * g[d] + b[d];
  if (relu) v = fmaxf(v, 0.f);
  Y[n * 256 + d] = v;
}

// global_add_pool: equal-size graphs -> contiguous 128-row sums
__global__ __launch_bounds__(256) void pool_kernel(const float* __restrict__ h,
                                                   float* __restrict__ out) {
  const int g = blockIdx.x, d = threadIdx.x;
  float s = 0.f;
  for (int rr = 0; rr < NPG; ++rr)
    s += h[((size_t)g * NPG + rr) * 256 + d];
  out[(size_t)g * 256 + d] = s;
}

// ---------------------------------------------------------------------------
// Host-side orchestration
// ---------------------------------------------------------------------------
extern "C" void kernel_launch(void* const* d_in, const int* in_sizes, int n_in,
                              void* d_out, int out_size, void* d_ws, size_t ws_size,
                              hipStream_t stream) {
  const int*   x        = (const int*)d_in[0];
  const int*   ei       = (const int*)d_in[1];        // [2, E]
  const int*   eattr    = (const int*)d_in[2];        // [E, 3]
  // d_in[3] = batch (implicit: equal-size graphs)
  const float* pe       = (const float*)d_in[4];
  const float* atom_emb = (const float*)d_in[5];
  const float* bond_emb = (const float*)d_in[6];
  const float* pe_w     = (const float*)d_in[7];
  const float* pe_b     = (const float*)d_in[8];
  const float* gine_w1  = (const float*)d_in[9];
  const float* gine_b1  = (const float*)d_in[10];
  const float* gine_g1  = (const float*)d_in[11];
  const float* gine_be1 = (const float*)d_in[12];
  const float* gine_w2  = (const float*)d_in[13];
  const float* gine_b2  = (const float*)d_in[14];
  const float* wqkv     = (const float*)d_in[15];
  const float* bqkv     = (const float*)d_in[16];
  const float* wo       = (const float*)d_in[17];
  const float* bo       = (const float*)d_in[18];
  const float* n1g      = (const float*)d_in[19];
  const float* n1b      = (const float*)d_in[20];
  const float* n2g      = (const float*)d_in[21];
  const float* n2b      = (const float*)d_in[22];
  const float* n3g      = (const float*)d_in[23];
  const float* n3b      = (const float*)d_in[24];
  const float* mw1      = (const float*)d_in[25];
  const float* mb1      = (const float*)d_in[26];
  const float* mw2      = (const float*)d_in[27];
  const float* mb2      = (const float*)d_in[28];

  const size_t ND = (size_t)Nn * Dd;                  // 64 MB fp32
  float* bufH   = (float*)d_ws;                       // layer input h
  float* bufA   = bufH + ND;                          // agg / z
  float* bufB   = bufA + ND;                          // pre-BN scratch
  float* bufC   = bufB + ND;                          // post-BN / h_local
  float* bufQKV = bufC + ND;                          // N x 768 (also MLP hidden)
  float* bufO   = bufQKV + (size_t)Nn * 768;          // attention out / h_attn
  float* stats  = bufO + ND;                          // 2 x 256 column stats

  auto gemm = [&](const float* A, const float* Wt, const float* bias,
                  const float* R, float* Y, int K, int M, int mode) {
    dim3 grid(Nn / 128, M / 64);
    wmma_gemm_kernel<<<grid, 256, 0, stream>>>(A, Wt, bias, R, Y, K, M, mode);
  };
  auto bnorm = [&](const float* T, const float* gv, const float* bv,
                   float* Yv, int relu) {
    (void)hipMemsetAsync(stats, 0, 2 * 256 * sizeof(float), stream);
    bn_stats_kernel<<<256, 256, 0, stream>>>(T, stats);
    bn_apply_kernel<<<Nn, 256, 0, stream>>>(T, stats, gv, bv, Yv, relu);
  };

  encode_kernel<<<Nn, 256, 0, stream>>>(x, pe, atom_emb, pe_w, pe_b, bufH);

  const size_t addBlocks = (ND + 255) / 256;
  for (int l = 0; l < Ll; ++l) {
    const float* w1  = gine_w1 + (size_t)l * Dd * Dd;
    const float* b1  = gine_b1 + (size_t)l * Dd;
    const float* g1  = gine_g1 + (size_t)l * Dd;
    const float* be1 = gine_be1 + (size_t)l * Dd;
    const float* w2  = gine_w2 + (size_t)l * Dd * Dd;
    const float* b2  = gine_b2 + (size_t)l * Dd;
    const float* wq  = wqkv + (size_t)l * Dd * 3 * Dd;
    const float* bq  = bqkv + (size_t)l * 3 * Dd;
    const float* wol = wo + (size_t)l * Dd * Dd;
    const float* bol = bo + (size_t)l * Dd;
    const float* w3  = mw1 + (size_t)l * Dd * FFf;
    const float* b3  = mb1 + (size_t)l * FFf;
    const float* w4  = mw2 + (size_t)l * FFf * Dd;
    const float* b4  = mb2 + (size_t)l * Dd;

    // --- GINEConv ---
    (void)hipMemsetAsync(bufA, 0, ND * sizeof(float), stream);
    message_kernel<<<Ee, 256, 0, stream>>>(bufH, ei, ei + Ee, eattr,
                                           bond_emb, bufA);
    add_kernel<<<addBlocks, 256, 0, stream>>>(bufA, bufH, ND);   // z = h + agg
    gemm(bufA, w1, b1, nullptr, bufB, Dd, Dd, 0);                // z @ w1 + b1
    bnorm(bufB, g1, be1, bufC, 1);                               // relu(BN(.))
    gemm(bufC, w2, b2, bufH, bufB, Dd, Dd, 2);                   // relu(.)+h
    bnorm(bufB, n1g + l * Dd, n1b + l * Dd, bufC, 0);            // h_local

    // --- global attention ---
    gemm(bufH, wq, bq, nullptr, bufQKV, Dd, 3 * Dd, 0);          // qkv
    attn_kernel<<<Gg * 8, 256, 0, stream>>>(bufQKV, bufO);       // o
    gemm(bufO, wol, bol, bufH, bufB, Dd, Dd, 3);                 // o@wo+bo+h
    bnorm(bufB, n2g + l * Dd, n2b + l * Dd, bufO, 0);            // h_attn

    // --- combine + MLP + norm3 ---
    add_kernel<<<addBlocks, 256, 0, stream>>>(bufC, bufO, ND);   // h_local+h_attn
    gemm(bufC, w3, b3, nullptr, bufQKV, Dd, FFf, 1);             // relu(.@w1+b1)
    gemm(bufQKV, w4, b4, nullptr, bufB, FFf, Dd, 0);             // .@w2+b2
    bnorm(bufB, n3g + l * Dd, n3b + l * Dd, bufH, 0);            // new h
  }

  pool_kernel<<<Gg, 256, 0, stream>>>(bufH, (float*)d_out);

  (void)in_sizes; (void)n_in; (void)out_size; (void)ws_size;
}
